// M2RNN_29111288332883
// MI455X (gfx1250) — matrix operand order repaired and checked by reading, not compiled, not run on hardware
//
#include <hip/hip_runtime.h>
#include <hip/hip_bf16.h>
#include <math.h>

typedef __attribute__((ext_vector_type(16))) __bf16 v16bf;
typedef __attribute__((ext_vector_type(8)))  float  v8f;
typedef __attribute__((ext_vector_type(2)))  float  v2f;

#define BB 4
#define SS 2048
#define DIN 2048
#define DOUT 2048
#define HH 8
#define KK 64
#define VV 64
#define NPROJ 1544   // H*K + H*K + H*V + H = 512+512+512+8

// Pack two fp32 -> packed bf16 pair (hardware cvt via native __bf16 casts).
__device__ inline unsigned pk_bf16(float x, float y) {
    __bf16 lo = (__bf16)x;
    __bf16 hi = (__bf16)y;
    unsigned short a, b;
    __builtin_memcpy(&a, &lo, 2);
    __builtin_memcpy(&b, &hi, 2);
    return (unsigned)a | ((unsigned)b << 16);
}

// ---------------------------------------------------------------------------
// Mixed-precision GEMM:  C(MxN) = A(MxKd) * W(KdxN), fp32 in/out, bf16 WMMA.
// Block tile 128(M) x 128(N), 256 threads = 8 waves; wave w owns rows w*16..+15
// and all 8 N-subtiles (one A fragment feeds 8 v_wmma per K-step).
// Double-buffered LDS + register-held global prefetch: loads for tile k+1 are
// issued before the WMMA block of tile k; one barrier per K-iteration.
// Cols >= sigCol get a sigmoid epilogue (forget gate).
// ---------------------------------------------------------------------------
__global__ __launch_bounds__(256)
void gemm_bf16_wmma(const float* __restrict__ A, const float* __restrict__ Wm,
                    float* __restrict__ C, int M, int N, int Kd, int sigCol)
{
    __shared__ unsigned short As[2][128][34];  // row-major bf16, padded
    __shared__ unsigned short Bt[2][128][34];  // transposed: Bt[n][k], padded

    const int t    = threadIdx.x;
    const int wv   = t >> 5;
    const int lane = t & 31;
    const int lc   = lane & 15;
    const bool hi  = lane >= 16;
    const int bm   = blockIdx.y * 128;
    const int bn   = blockIdx.x * 128;

    // Per-thread staging coordinates.
    const int am  = t >> 1;          // A row 0..127
    const int ak  = (t & 1) * 16;    // A col base {0,16}
    const int bk  = (t >> 4) * 2;    // W k-row pair base 0..30
    const int bn0 = (t & 15) * 8;    // W col base 0..120

    float4 ar[4], br0[2], br1[2];

    auto load_tiles = [&](int kk) {
        const float* ap = A + (long long)(bm + am) * Kd + kk + ak;
        #pragma unroll
        for (int j = 0; j < 4; ++j) ar[j] = *(const float4*)(ap + 4 * j);
        const float* bp0 = Wm + (long long)(kk + bk) * N;
        const float* bp1 = bp0 + N;
        #pragma unroll
        for (int j = 0; j < 2; ++j) {
            int gn = bn + bn0 + 4 * j;
            if (gn + 3 < N) {
                br0[j] = *(const float4*)(bp0 + gn);
                br1[j] = *(const float4*)(bp1 + gn);
            } else {
                float4 z0, z1;
                z0.x = (gn + 0 < N) ? bp0[gn + 0] : 0.f;
                z0.y = (gn + 1 < N) ? bp0[gn + 1] : 0.f;
                z0.z = (gn + 2 < N) ? bp0[gn + 2] : 0.f;
                z0.w = (gn + 3 < N) ? bp0[gn + 3] : 0.f;
                z1.x = (gn + 0 < N) ? bp1[gn + 0] : 0.f;
                z1.y = (gn + 1 < N) ? bp1[gn + 1] : 0.f;
                z1.z = (gn + 2 < N) ? bp1[gn + 2] : 0.f;
                z1.w = (gn + 3 < N) ? bp1[gn + 3] : 0.f;
                br0[j] = z0; br1[j] = z1;
            }
        }
    };
    auto store_tiles = [&](int buf) {
        unsigned* arow32 = (unsigned*)&As[buf][am][ak];   // even half-offset: aligned
        #pragma unroll
        for (int j = 0; j < 4; ++j) {
            arow32[2 * j + 0] = pk_bf16(ar[j].x, ar[j].y);
            arow32[2 * j + 1] = pk_bf16(ar[j].z, ar[j].w);
        }
        #pragma unroll
        for (int j = 0; j < 2; ++j) {
            const float* r0 = (const float*)&br0[j];
            const float* r1 = (const float*)&br1[j];
            #pragma unroll
            for (int e = 0; e < 4; ++e)   // pack k-pair (bk, bk+1) per column
                *(unsigned*)&Bt[buf][bn0 + 4 * j + e][bk] = pk_bf16(r0[e], r1[e]);
        }
    };

    v8f acc[8] = {};
    const int nK = Kd / 32;

    load_tiles(0);
    store_tiles(0);
    __syncthreads();

    int cur = 0;
    for (int kt = 0; kt < nK; ++kt) {
        if (kt + 1 < nK) load_tiles((kt + 1) * 32);   // issue global loads early

        // ---- A fragment: 16x32 bf16 layout (ISA 7.12.2) --------------------
        union { v16bf v; unsigned u[8]; } afr;
        const int arow  = wv * 16 + lc;
        const int kbase = hi ? 8 : 0;
        #pragma unroll
        for (int j = 0; j < 4; ++j) {
            afr.u[j]     = *(const unsigned*)&As[cur][arow][kbase + 2 * j];
            afr.u[4 + j] = *(const unsigned*)&As[cur][arow][16 + kbase + 2 * j];
        }
        // ---- 8 N-subtiles: B fragment + WMMA -------------------------------
        #pragma unroll
        for (int nt = 0; nt < 8; ++nt) {
            union { v16bf v; unsigned u[8]; } bfr;
            const int bcol  = nt * 16 + lc;
            const int rbase = hi ? 8 : 0;
            #pragma unroll
            for (int j = 0; j < 4; ++j) {
                bfr.u[j]     = *(const unsigned*)&Bt[cur][bcol][rbase + 2 * j];
                bfr.u[4 + j] = *(const unsigned*)&Bt[cur][bcol][16 + rbase + 2 * j];
            }
            acc[nt] = __builtin_amdgcn_wmma_f32_16x16x32_bf16(
                false, afr.v, false, bfr.v, (short)0, acc[nt], false, false);
        }

        if (kt + 1 < nK) store_tiles(cur ^ 1);  // waits loads, fills other buffer
        __syncthreads();                        // orders reads(cur) vs next writes
        cur ^= 1;
    }

    // ---- epilogue: C/D layout VGPR r -> M = r (+8 for hi half) -------------
    #pragma unroll
    for (int nt = 0; nt < 8; ++nt) {
        #pragma unroll
        for (int r = 0; r < 8; ++r) {
            int row = bm + wv * 16 + r + (hi ? 8 : 0);
            int col = bn + nt * 16 + lc;
            if (col < N) {
                float vres = acc[nt][r];
                if (col >= sigCol) vres = 1.0f / (1.0f + __expf(-vres));
                C[(long long)row * N + col] = vres;
            }
        }
    }
}

// ---------------------------------------------------------------------------
// Matrix-state recurrence: one block per (b,h) chain (32 blocks total).
// state(64x64, fp32 in LDS):  state = f * (state @ SW_h) + k v^T ; y = q^T state
// state @ SW_h via v_wmma_f32_16x16x4_f32: 8 waves, wave w owns M-tile w/2 and
// 2 of 4 N-tiles; SW fragments loop-invariant in VGPRs for the whole scan.
// proj (50 MB) is L2-resident (192 MB), so per-step q/k/v/f reads are L2 hits.
// ---------------------------------------------------------------------------
__global__ __launch_bounds__(256)
void scan_kernel(const float* __restrict__ proj, const float* __restrict__ state_w,
                 const float* __restrict__ init_state, float* __restrict__ ys,
                 float* __restrict__ final_state)
{
    __shared__ float st[64][65];            // state, padded (conflict-free)
    __shared__ float qb[64], kb[64], vb[64];

    const int t    = threadIdx.x;
    const int wv   = t >> 5;
    const int lane = t & 31;
    const int lc   = lane & 15;
    const bool hi  = lane >= 16;
    const int bh   = blockIdx.x;            // b*H + h
    const int h    = bh & (HH - 1);
    const int mt   = wv >> 1;               // M-tile 0..3
    const int ntb  = (wv & 1) * 2;          // first of this wave's 2 N-tiles

    for (int i = t; i < KK * VV; i += 256)
        st[i >> 6][i & 63] = init_state[(long long)bh * (KK * VV) + i];

    // Loop-invariant SW B-fragments (f32 4x16 layout: V0 -> rows {0,2}, V1 -> {1,3})
    v2f swb[16][2];
    const float* SW = state_w + h * (VV * VV);   // SW[v][w], row-major 64
    #pragma unroll
    for (int c = 0; c < 16; ++c) {
        #pragma unroll
        for (int l = 0; l < 2; ++l) {
            int col = (ntb + l) * 16 + lc;
            int r0  = 4 * c + (hi ? 2 : 0);
            swb[c][l].x = SW[(r0 + 0) * 64 + col];
            swb[c][l].y = SW[(r0 + 1) * 64 + col];
        }
    }
    const v8f vzero = {0.f, 0.f, 0.f, 0.f, 0.f, 0.f, 0.f, 0.f};
    __syncthreads();

    const long long rowBase = (long long)(bh >> 3) * SS;   // b*S
    for (int s = 0; s < SS; ++s) {
        const float* p = proj + (rowBase + s) * NPROJ;

        __syncthreads();                    // prior readout done before q/k/v overwrite
        if (t < 64) {
            qb[t] = p[h * KK + t];
            kb[t] = p[HH * KK + h * KK + t];
            vb[t] = p[2 * HH * KK + h * VV + t];
        }
        const float f = p[2 * HH * KK + HH * VV + h];  // sigmoid already applied
        if (s + 1 < SS && t < 6) {          // prefetch next step's q/k/v rows
            int region = t >> 1, off = (t & 1) * 32;
            int basec = (region == 0) ? h * KK
                      : (region == 1) ? HH * KK + h * KK
                                      : 2 * HH * KK + h * VV;
            __builtin_prefetch(p + NPROJ + basec + off, 0, 3);
        }

        // decayed = state @ SW  (fp32 WMMA, K-chunks of 4)
        v8f acc[2]; acc[0] = vzero; acc[1] = vzero;
        const int arow = mt * 16 + lc;
        #pragma unroll
        for (int c = 0; c < 16; ++c) {
            v2f a;
            const int k0 = 4 * c + (hi ? 2 : 0);
            a.x = st[arow][k0];
            a.y = st[arow][k0 + 1];
            acc[0] = __builtin_amdgcn_wmma_f32_16x16x4_f32(
                false, a, false, swb[c][0], (short)0, acc[0], false, false);
            acc[1] = __builtin_amdgcn_wmma_f32_16x16x4_f32(
                false, a, false, swb[c][1], (short)0, acc[1], false, false);
        }
        __syncthreads();                    // all reads of old state complete

        // state = f * decayed + k v^T
        #pragma unroll
        for (int l = 0; l < 2; ++l) {
            const int col = (ntb + l) * 16 + lc;
            const float vcol = vb[col];
            #pragma unroll
            for (int r = 0; r < 8; ++r) {
                const int row = mt * 16 + r + (hi ? 8 : 0);
                st[row][col] = f * acc[l][r] + kb[row] * vcol;
            }
        }
        __syncthreads();                    // new state visible

        // y = q^T state  (64-wide matvec on VALU)
        if (t < 64) {
            float y = 0.f;
            #pragma unroll 8
            for (int ki = 0; ki < KK; ++ki)
                y = fmaf(qb[ki], st[ki][t], y);
            ys[(rowBase + s) * (HH * VV) + h * VV + t] = y;
        }
    }

    __syncthreads();
    for (int i = t; i < KK * VV; i += 256)
        final_state[(long long)bh * (KK * VV) + i] = st[i >> 6][i & 63];
}

extern "C" void kernel_launch(void* const* d_in, const int* in_sizes, int n_in,
                              void* d_out, int out_size, void* d_ws, size_t ws_size,
                              hipStream_t stream) {
    const float* x            = (const float*)d_in[0];
    const float* input_state  = (const float*)d_in[1];
    const float* w_in         = (const float*)d_in[2];
    const float* state_weight = (const float*)d_in[3];
    const float* w_out        = (const float*)d_in[4];

    float* out         = (float*)d_out;                                  // (B,S,DOUT)
    float* final_state = out + (long long)BB * SS * DOUT;                // (B,H,K*V)
    float* proj        = (float*)d_ws;                                   // (B*S, NPROJ)
    float* ysb         = proj + (long long)BB * SS * NPROJ;              // (B*S, H*V)

    const int M = BB * SS;  // 8192

    // 1) proj = x @ w_in, sigmoid on forget-gate columns (>= 1536)
    gemm_bf16_wmma<<<dim3((NPROJ + 127) / 128, M / 128), 256, 0, stream>>>(
        x, w_in, proj, M, NPROJ, DIN, 2 * HH * KK + HH * VV);

    // 2) sequential matrix-state scan, one block per (b,h)
    scan_kernel<<<BB * HH, 256, 0, stream>>>(proj, state_weight, input_state,
                                             ysb, final_state);

    // 3) out = ys @ w_out
    gemm_bf16_wmma<<<dim3(DOUT / 128, M / 128), 256, 0, stream>>>(
        ysb, w_out, out, M, DOUT, HH * VV, 0x7fffffff);
}
